// MRKGCN_88347477278835
// MI455X (gfx1250) — matrix operand order repaired and checked by reading, not compiled, not run on hardware
//
#include <hip/hip_runtime.h>
#include <hip/hip_bf16.h>

typedef __bf16 bf16;
typedef __attribute__((ext_vector_type(16))) __bf16 v16bf;
typedef __attribute__((ext_vector_type(8)))  __bf16 v8bf;
typedef __attribute__((ext_vector_type(8)))  float  v8f;

#define NN 8000
#define TT 12
#define EE 256000
#define RR 34
#define NBB 4
#define DD 64
#define DUU 32
#define FFF 2048

// ---------------------------------------------------------------------------
// WMMA helpers (CDNA5 wave32, 16x16x32 bf16 -> f32)
// ---------------------------------------------------------------------------
__device__ __forceinline__ v8f wmma_bf16(v16bf a, v16bf b, v8f c) {
  return __builtin_amdgcn_wmma_f32_16x16x32_bf16(false, a, false, b, (short)0, c, false, false);
}

// A-operand (16x32 bf16) per-lane load: lane half h holds K chunks [8h..8h+7]
// (elements 0..7) and [16+8h..16+8h+7] (elements 8..15) of its row.
__device__ __forceinline__ v16bf load_a16(const bf16* row, int half) {
  v8bf lo = *(const v8bf*)(row + 8 * half);
  v8bf hi = *(const v8bf*)(row + 16 + 8 * half);
  v16bf r;
#pragma unroll
  for (int i = 0; i < 8; ++i) { r[i] = lo[i]; r[i + 8] = hi[i]; }
  return r;
}

// ---------------------------------------------------------------------------
// Weight packing into WMMA B layout: dst[((kb*NT+nt)*32+lane)*16+e],
// n = nt*16 + lane%16, k = kb*32 + 8*(lane/16) + (e%8) + 16*(e/8)
// ---------------------------------------------------------------------------
__global__ void k_pack_b(const float* __restrict__ src, bf16* __restrict__ dst,
                         int KB, int NT, int trans) {
  long idx = (long)blockIdx.x * blockDim.x + threadIdx.x;
  long total = (long)KB * NT * 512;
  if (idx >= total) return;
  int e = idx & 15;
  int lane = (idx >> 4) & 31;
  int nt = (int)((idx >> 9) % NT);
  int kb = (int)(idx / ((long)NT * 512));
  int n = nt * 16 + (lane & 15);
  int half = lane >> 4;
  int k = kb * 32 + 8 * half + (e & 7) + 16 * (e >> 3);
  int K = KB * 32, Ncols = NT * 16;
  float v = trans ? src[(long)n * K + k] : src[(long)k * Ncols + n];
  dst[idx] = (bf16)v;
}

// Basis-combined relation weights W_r = sum_b comb[r,b]*bases[b], packed per r.
__global__ void k_pack_basis(const float* __restrict__ comb, const float* __restrict__ bases,
                             bf16* __restrict__ dst, int Din, int KB, int NT) {
  long idx = (long)blockIdx.x * blockDim.x + threadIdx.x;
  long per = (long)KB * NT * 512;
  if (idx >= (long)RR * per) return;
  int r = (int)(idx / per);
  long rem = idx % per;
  int e = rem & 15;
  int lane = (rem >> 4) & 31;
  int nt = (int)((rem >> 9) % NT);
  int kb = (int)(rem / ((long)NT * 512));
  int n = nt * 16 + (lane & 15);
  int half = lane >> 4;
  int k = kb * 32 + 8 * half + (e & 7) + 16 * (e >> 3);
  float v = 0.f;
#pragma unroll
  for (int b = 0; b < NBB; ++b) v += comb[r * NBB + b] * bases[((long)b * Din + k) * Din + n];
  dst[idx] = (bf16)v;
}

// ---------------------------------------------------------------------------
// Generic GEMM, compile-time specialized on (KB, NT) so the inner loop is a
// straight global_load_b128 -> v_wmma stream with no per-tile branching.
// C[M x NT*16] = A[M x KB*32](bf16) @ Bpack (+bias), batched over blockIdx.y.
// ---------------------------------------------------------------------------
template <int KB, int NT>
__global__ void __launch_bounds__(128)
k_gemm(const bf16* __restrict__ A, const bf16* __restrict__ Bp,
       const float* __restrict__ bias,
       float* __restrict__ C, int ldc,
       bf16* __restrict__ Cbf, int ldcb,
       int M, long strideB, long strideC) {
  int batch = blockIdx.y;
  int lane = threadIdx.x & 31, warp = threadIdx.x >> 5;
  int mt = blockIdx.x * 4 + warp;
  if (mt >= (M >> 4)) return;
  int half = lane >> 4, mloc = lane & 15;
  constexpr int K = KB * 32;
  const bf16* Bb = Bp + batch * strideB;
  float* Cb = C + batch * strideC;
  long row = (long)mt * 16 + mloc;
  v16bf a[KB];
#pragma unroll
  for (int kb = 0; kb < KB; ++kb) a[kb] = load_a16(A + row * K + kb * 32, half);
#pragma unroll
  for (int nt = 0; nt < NT; ++nt) {
    v8f acc = {};
#pragma unroll
    for (int kb = 0; kb < KB; ++kb) {
      v16bf b = *(const v16bf*)(Bb + (((long)kb * NT + nt) * 32 + lane) * 16);
      acc = wmma_bf16(a[kb], b, acc);
    }
    int col = nt * 16 + mloc;
    float bs = bias ? bias[col] : 0.f;
#pragma unroll
    for (int e = 0; e < 8; ++e) {
      long r = (long)mt * 16 + e + 8 * half;
      float v = acc[e] + bs;
      Cb[r * ldc + col] = v;
      if (Cbf) Cbf[r * ldcb + col] = (bf16)v;
    }
  }
}

// ---------------------------------------------------------------------------
// Causal conv 1->64 (k=3, left pad 2) + tanh, writes h_bf layout (t, n, d)
// ---------------------------------------------------------------------------
__global__ void k_conv(const float* __restrict__ x, const float* __restrict__ cw,
                       const float* __restrict__ cb, bf16* __restrict__ hbf) {
  long idx = (long)blockIdx.x * blockDim.x + threadIdx.x;
  if (idx >= (long)NN * TT * DD) return;
  int d = idx & 63;
  long r = idx >> 6;
  int t = (int)(r % TT);
  long n = r / TT;
  float s = cb[d];
#pragma unroll
  for (int j = 0; j < 3; ++j) {
    int tt2 = t - 2 + j;
    if (tt2 >= 0) s += x[n * TT + tt2] * cw[d * 3 + j];
  }
  hbf[((long)t * NN + n) * DD + d] = (bf16)tanhf(s);
}

// ---------------------------------------------------------------------------
// Edge gather + atomic scatter: dest[dst] += proj[et, src]
// ---------------------------------------------------------------------------
__global__ void k_edge(const float* __restrict__ proj, const int* __restrict__ src,
                       const int* __restrict__ dst, const int* __restrict__ et,
                       float* __restrict__ dest, int ldd, int Dd) {
  long idx = (long)blockIdx.x * blockDim.x + threadIdx.x;
  int per = Dd >> 2;
  long e = idx / per;
  if (e >= EE) return;
  int q = (int)(idx % per) * 4;
  int r = et[e], s = src[e], dd = dst[e];
  const float4 v = *(const float4*)(proj + ((long)r * NN + s) * Dd + q);
  float* p = dest + (long)dd * ldd + q;
  atomicAdd(p + 0, v.x); atomicAdd(p + 1, v.y);
  atomicAdd(p + 2, v.z); atomicAdd(p + 3, v.w);
}

// Sinusoidal positional encoding add, plus bf16 shadow
__global__ void k_pe(float* __restrict__ x1, bf16* __restrict__ x1bf) {
  long idx = (long)blockIdx.x * blockDim.x + threadIdx.x;
  if (idx >= (long)NN * TT * DD) return;
  int d = idx & 63;
  long r = idx >> 6;
  int t = (int)(r % TT);
  int i2 = (d >> 1) << 1;
  float freq = __expf(-(float)i2 * (9.210340371976184f / 64.f));
  float ang = (float)t * freq;
  float pe = (d & 1) ? __cosf(ang) : __sinf(ang);
  float v = x1[idx] + pe;
  x1[idx] = v;
  x1bf[idx] = (bf16)v;
}

// Per-node multi-head attention over T=12 (8 heads, dh=8), tiny VALU kernel
__global__ void k_attn(const float* __restrict__ qkv, float* __restrict__ a,
                       bf16* __restrict__ abf) {
  __shared__ float sq[TT][DD], sk[TT][DD], sv[TT][DD];
  int n = blockIdx.x, tid = threadIdx.x;
  for (int idx = tid; idx < TT * 192; idx += 96) {
    int t = idx / 192, c = idx % 192;
    float v = qkv[((long)n * TT + t) * 192 + c];
    if (c < 64) sq[t][c] = v;
    else if (c < 128) sk[t][c - 64] = v;
    else sv[t][c - 128] = v;
  }
  __syncthreads();
  int t = tid % TT, h = tid / TT;
  if (h < 8) {
    int base = h * 8;
    float s[TT], mx = -1e30f;
    for (int u = 0; u < TT; ++u) {
      float d = 0.f;
#pragma unroll
      for (int q = 0; q < 8; ++q) d += sq[t][base + q] * sk[u][base + q];
      s[u] = d * 0.3535533905932738f;  // 1/sqrt(8)
      mx = fmaxf(mx, s[u]);
    }
    float sum = 0.f;
    for (int u = 0; u < TT; ++u) { s[u] = __expf(s[u] - mx); sum += s[u]; }
    float inv = 1.f / sum;
#pragma unroll
    for (int q = 0; q < 8; ++q) {
      float o = 0.f;
      for (int u = 0; u < TT; ++u) o += s[u] * sv[u][base + q];
      o *= inv;
      long oi = ((long)n * TT + t) * DD + base + q;
      a[oi] = o;
      abf[oi] = (bf16)o;
    }
  }
}

// Residual + LayerNorm over 64 columns, one warp per row
__global__ void k_lnres(const float* __restrict__ X, const float* __restrict__ Rr,
                        const float* __restrict__ w, const float* __restrict__ b,
                        float* __restrict__ Y, bf16* __restrict__ Ybf, int M) {
  int warp = threadIdx.x >> 5, lane = threadIdx.x & 31;
  long row = (long)blockIdx.x * 8 + warp;
  if (row >= M) return;
  const float* xp = X + row * DD;
  const float* rp = Rr + row * DD;
  float a0 = xp[lane] + rp[lane];
  float a1 = xp[lane + 32] + rp[lane + 32];
  float s = a0 + a1, s2 = a0 * a0 + a1 * a1;
#pragma unroll
  for (int off = 16; off >= 1; off >>= 1) {
    s += __shfl_xor(s, off, 32);
    s2 += __shfl_xor(s2, off, 32);
  }
  float mean = s * (1.f / 64.f);
  float var = s2 * (1.f / 64.f) - mean * mean;
  float rstd = rsqrtf(var + 1e-5f);
  float y0 = (a0 - mean) * rstd * w[lane] + b[lane];
  float y1 = (a1 - mean) * rstd * w[lane + 32] + b[lane + 32];
  Y[row * DD + lane] = y0;
  Y[row * DD + lane + 32] = y1;
  if (Ybf) {
    Ybf[row * DD + lane] = (bf16)y0;
    Ybf[row * DD + lane + 32] = (bf16)y1;
  }
}

// ---------------------------------------------------------------------------
// Fused FFN 64 -> 2048 (relu) -> 64, hidden kept in regs/LDS (one wave = 16 rows)
// ---------------------------------------------------------------------------
__global__ void __launch_bounds__(128)
k_ffn(const bf16* __restrict__ X, const bf16* __restrict__ W1p,
      const float* __restrict__ b1, const bf16* __restrict__ W2p,
      const float* __restrict__ b2, float* __restrict__ F) {
  __shared__ __align__(16) bf16 stg[4][16][32];
  int lane = threadIdx.x & 31, warp = threadIdx.x >> 5;
  int mt = blockIdx.x * 4 + warp;
  int half = lane >> 4, mloc = lane & 15;
  long row = (long)mt * 16 + mloc;
  v16bf a0 = load_a16(X + row * DD, half);
  v16bf a1 = load_a16(X + row * DD + 32, half);
  v8f acc2[4];
#pragma unroll
  for (int i = 0; i < 4; ++i) { v8f z = {}; acc2[i] = z; }
  for (int hb = 0; hb < 64; ++hb) {
#pragma unroll
    for (int j = 0; j < 2; ++j) {
      int nt = hb * 2 + j;
      v8f acc = {};
      v16bf bA = *(const v16bf*)(W1p + (((long)0 * 128 + nt) * 32 + lane) * 16);
      acc = wmma_bf16(a0, bA, acc);
      v16bf bB = *(const v16bf*)(W1p + (((long)1 * 128 + nt) * 32 + lane) * 16);
      acc = wmma_bf16(a1, bB, acc);
      float bs = b1[nt * 16 + mloc];
#pragma unroll
      for (int e = 0; e < 8; ++e) {
        float v = acc[e] + bs;
        v = v > 0.f ? v : 0.f;
        stg[warp][e + 8 * half][j * 16 + mloc] = (bf16)v;
      }
    }
    __syncthreads();
    v16bf ah = load_a16(&stg[warp][mloc][0], half);
#pragma unroll
    for (int nt = 0; nt < 4; ++nt) {
      v16bf b = *(const v16bf*)(W2p + (((long)hb * 4 + nt) * 32 + lane) * 16);
      acc2[nt] = wmma_bf16(ah, b, acc2[nt]);
    }
    __syncthreads();
  }
#pragma unroll
  for (int nt = 0; nt < 4; ++nt) {
    int col = nt * 16 + mloc;
    float bs = b2[col];
#pragma unroll
    for (int e = 0; e < 8; ++e) {
      long r = (long)mt * 16 + e + 8 * half;
      F[r * DD + col] = acc2[nt][e] + bs;
    }
  }
}

// Mean over T -> first 64 cols of c (bf16); urban concat fills cols 64..95
__global__ void k_mean(const float* __restrict__ x2, bf16* __restrict__ cbf) {
  long idx = (long)blockIdx.x * blockDim.x + threadIdx.x;
  if (idx >= (long)NN * DD) return;
  int d = idx & 63;
  long n = idx >> 6;
  float s = 0.f;
#pragma unroll
  for (int t = 0; t < TT; ++t) s += x2[(n * TT + t) * DD + d];
  cbf[n * 96 + d] = (bf16)(s * (1.f / 12.f));
}

__global__ void k_concat_u(const float* __restrict__ hu, bf16* __restrict__ cbf) {
  long idx = (long)blockIdx.x * blockDim.x + threadIdx.x;
  if (idx >= (long)NN * DUU) return;
  int d = idx & 31;
  long n = idx >> 5;
  cbf[n * 96 + 64 + d] = (bf16)hu[n * DUU + d];
}

__global__ void k_cvt(const float* __restrict__ src, bf16* __restrict__ dst, long n) {
  long idx = (long)blockIdx.x * blockDim.x + threadIdx.x;
  if (idx < n) dst[idx] = (bf16)src[idx];
}

// ---------------------------------------------------------------------------
// Global 8000x8000 attention, flash-style: one wave per 16 queries.
// ---------------------------------------------------------------------------
__global__ void __launch_bounds__(32)
k_gattn(const bf16* __restrict__ Qbf, const bf16* __restrict__ Kp,
        const bf16* __restrict__ Vp, const float* __restrict__ ow,
        const float* __restrict__ ob, float* __restrict__ out) {
  __shared__ __align__(16) bf16 stg[16][32];
  int lane = threadIdx.x & 31;
  int half = lane >> 4, mloc = lane & 15;
  int qb = blockIdx.x;
  long row = (long)qb * 16 + mloc;
  v16bf aq = load_a16(Qbf + row * 32, half);
  float mrun[8], lrun[8];
  v8f acco = {};
#pragma unroll
  for (int e = 0; e < 8; ++e) { mrun[e] = -1e30f; lrun[e] = 0.f; }
  const float sc = 0.17677669529663687f;  // 1/sqrt(32)
  for (int kb = 0; kb < 250; ++kb) {
    v16bf b0 = *(const v16bf*)(Kp + ((long)(2 * kb) * 32 + lane) * 16);
    v16bf b1 = *(const v16bf*)(Kp + ((long)(2 * kb + 1) * 32 + lane) * 16);
    v8f z = {};
    v8f s0 = wmma_bf16(aq, b0, z);
    v8f s1 = wmma_bf16(aq, b1, z);
#pragma unroll
    for (int e = 0; e < 8; ++e) {
      float x0 = s0[e] * sc, x1v = s1[e] * sc;
      float mx = fmaxf(x0, x1v);
#pragma unroll
      for (int off = 8; off >= 1; off >>= 1) mx = fmaxf(mx, __shfl_xor(mx, off, 16));
      float mnew = fmaxf(mrun[e], mx);
      float corr = __expf(mrun[e] - mnew);
      float p0 = __expf(x0 - mnew);
      float p1 = __expf(x1v - mnew);
      float rs = p0 + p1;
#pragma unroll
      for (int off = 8; off >= 1; off >>= 1) rs += __shfl_xor(rs, off, 16);
      lrun[e] = lrun[e] * corr + rs;
      mrun[e] = mnew;
      acco[e] *= corr;
      stg[e + 8 * half][mloc] = (bf16)p0;
      stg[e + 8 * half][16 + mloc] = (bf16)p1;
    }
    __syncthreads();
    v16bf ap = load_a16(&stg[mloc][0], half);
    v16bf bv = *(const v16bf*)(Vp + ((long)kb * 32 + lane) * 16);
    acco = wmma_bf16(ap, bv, acco);
    __syncthreads();
  }
#pragma unroll
  for (int e = 0; e < 8; ++e) {
    float o = acco[e] / lrun[e];
    float v = o * ow[mloc];
#pragma unroll
    for (int off = 8; off >= 1; off >>= 1) v += __shfl_xor(v, off, 16);
    if (mloc == 0) out[qb * 16 + e + 8 * half] = v + ob[0];
  }
}

// ---------------------------------------------------------------------------
// Host launcher
// ---------------------------------------------------------------------------
extern "C" void kernel_launch(void* const* d_in, const int* in_sizes, int n_in,
                              void* d_out, int out_size, void* d_ws, size_t ws_size,
                              hipStream_t stream) {
  const float* x        = (const float*)d_in[0];
  const float* x_urban  = (const float*)d_in[1];
  const int* src_base   = (const int*)d_in[2];
  const int* dst_base   = (const int*)d_in[3];
  const int* et_base    = (const int*)d_in[4];
  const int* src_urb    = (const int*)d_in[5];
  const int* dst_urb    = (const int*)d_in[6];
  const int* et_urb     = (const int*)d_in[7];
  const float* conv_w   = (const float*)d_in[8];
  const float* conv_b   = (const float*)d_in[9];
  const float* comb_b   = (const float*)d_in[10];
  const float* bases_b  = (const float*)d_in[11];
  const float* wself_b  = (const float*)d_in[12];
  const float* gbias_b  = (const float*)d_in[13];
  const float* in_w     = (const float*)d_in[14];
  const float* in_b     = (const float*)d_in[15];
  const float* outp_w   = (const float*)d_in[16];
  const float* outp_b   = (const float*)d_in[17];
  const float* ln1_w    = (const float*)d_in[18];
  const float* ln1_b    = (const float*)d_in[19];
  const float* ln2_w    = (const float*)d_in[20];
  const float* ln2_b    = (const float*)d_in[21];
  const float* ff1_w    = (const float*)d_in[22];
  const float* ff1_b    = (const float*)d_in[23];
  const float* ff2_w    = (const float*)d_in[24];
  const float* ff2_b    = (const float*)d_in[25];
  const float* comb_u   = (const float*)d_in[26];
  const float* bases_u  = (const float*)d_in[27];
  const float* wself_u  = (const float*)d_in[28];
  const float* gbias_u  = (const float*)d_in[29];
  const float* wq       = (const float*)d_in[30];
  const float* wk       = (const float*)d_in[31];
  const float* wv       = (const float*)d_in[32];
  const float* out_w    = (const float*)d_in[33];
  const float* out_b    = (const float*)d_in[34];
  float* out = (float*)d_out;

  char* W = (char*)d_ws;
  size_t off = 0;
  auto carve = [&](size_t bytes) -> void* {
    off = (off + 255) & ~(size_t)255;
    void* p = W + off;
    off += bytes;
    return p;
  };
  const size_t NT64 = (size_t)NN * TT * DD;  // 6.144M elements
  bf16* h_bf   = (bf16*)carve(NT64 * 2);
  bf16* wbp    = (bf16*)carve((size_t)RR * 4096 * 2);
  bf16* wsp    = (bf16*)carve(4096 * 2);
  bf16* inwp   = (bf16*)carve((size_t)2 * 12 * 512 * 2);
  bf16* outwp  = (bf16*)carve(4096 * 2);
  bf16* ff1p   = (bf16*)carve((size_t)2 * 128 * 512 * 2);
  bf16* ff2p   = (bf16*)carve((size_t)64 * 4 * 512 * 2);
  bf16* wup    = (bf16*)carve((size_t)RR * 1024 * 2);
  bf16* wsup   = (bf16*)carve(1024 * 2);
  bf16* wqp    = (bf16*)carve((size_t)3 * 2 * 512 * 2);
  bf16* wkp    = (bf16*)carve((size_t)3 * 2 * 512 * 2);
  bf16* wvp    = (bf16*)carve((size_t)3 * 1 * 512 * 2);
  float* proj  = (float*)carve((size_t)RR * NN * DD * 4);
  float* x1    = (float*)carve(NT64 * 4);
  bf16* x1bf   = (bf16*)carve(NT64 * 2);
  float* qkv   = (float*)carve(NT64 * 3 * 4);   // region reused after attention
  float* a     = (float*)carve(NT64 * 4);
  bf16* abf    = (bf16*)carve(NT64 * 2);
  float* x1ln  = (float*)carve(NT64 * 4);
  bf16* x1lnbf = (bf16*)carve(NT64 * 2);
  float* aout = qkv;                 // overlay: qkv dead after k_attn
  float* f    = qkv + NT64;
  float* x2   = qkv + 2 * NT64;
  bf16* xubf   = (bf16*)carve((size_t)NN * DUU * 2);
  float* hu    = (float*)carve((size_t)NN * DUU * 4);
  bf16* cbf    = (bf16*)carve((size_t)NN * 96 * 2);
  float* q2f   = (float*)carve((size_t)NN * 32 * 4);
  bf16* q2bf   = (bf16*)carve((size_t)NN * 32 * 2);
  float* k2    = (float*)carve((size_t)NN * 32 * 4);
  float* v2    = (float*)carve((size_t)NN * 16 * 4);
  bf16* k2p    = (bf16*)carve((size_t)500 * 512 * 2);
  bf16* v2p    = (bf16*)carve((size_t)250 * 512 * 2);

  // ---- weight packs + embeddings ----
  k_pack_basis<<<(RR * 4096 + 255) / 256, 256, 0, stream>>>(comb_b, bases_b, wbp, 64, 2, 4);
  k_pack_b<<<16, 256, 0, stream>>>(wself_b, wsp, 2, 4, 0);
  k_pack_b<<<48, 256, 0, stream>>>(in_w, inwp, 2, 12, 1);
  k_pack_b<<<16, 256, 0, stream>>>(outp_w, outwp, 2, 4, 1);
  k_pack_b<<<512, 256, 0, stream>>>(ff1_w, ff1p, 2, 128, 1);
  k_pack_b<<<512, 256, 0, stream>>>(ff2_w, ff2p, 64, 4, 1);
  k_pack_basis<<<(RR * 1024 + 255) / 256, 256, 0, stream>>>(comb_u, bases_u, wup, 32, 1, 2);
  k_pack_b<<<4, 256, 0, stream>>>(wself_u, wsup, 1, 2, 0);
  k_pack_b<<<12, 256, 0, stream>>>(wq, wqp, 3, 2, 1);
  k_pack_b<<<12, 256, 0, stream>>>(wk, wkp, 3, 2, 1);
  k_pack_b<<<6, 256, 0, stream>>>(wv, wvp, 3, 1, 1);
  k_conv<<<(int)(NT64 / 256), 256, 0, stream>>>(x, conv_w, conv_b, h_bf);
  k_cvt<<<(NN * DUU + 255) / 256, 256, 0, stream>>>(x_urban, xubf, (long)NN * DUU);

  // ---- per-timestep basis RGCN on base graph ----
  for (int t = 0; t < TT; ++t) {
    const bf16* ht = h_bf + (size_t)t * NN * DD;
    // self-loop + bias initializes x1 rows (n*T+t)
    k_gemm<2, 4><<<dim3(125, 1), 128, 0, stream>>>(ht, wsp, gbias_b, x1 + t * DD,
                                                   TT * DD, nullptr, 0, NN, 0, 0);
    // per-relation projections
    k_gemm<2, 4><<<dim3(125, RR), 128, 0, stream>>>(ht, wbp, nullptr, proj, DD,
                                                    nullptr, 0, NN, 4096, (long)NN * DD);
    // gather/scatter edges
    k_edge<<<16000, 256, 0, stream>>>(proj, src_base, dst_base, et_base,
                                      x1 + t * DD, TT * DD, DD);
  }
  // ---- positional encoding + transformer layer ----
  k_pe<<<(int)(NT64 / 256), 256, 0, stream>>>(x1, x1bf);
  k_gemm<2, 12><<<dim3(1500, 1), 128, 0, stream>>>(x1bf, inwp, in_b, qkv, 192,
                                                   nullptr, 0, NN * TT, 0, 0);
  k_attn<<<NN, 96, 0, stream>>>(qkv, a, abf);
  k_gemm<2, 4><<<dim3(1500, 1), 128, 0, stream>>>(abf, outwp, outp_b, aout, DD,
                                                  nullptr, 0, NN * TT, 0, 0);
  k_lnres<<<12000, 256, 0, stream>>>(x1, aout, ln1_w, ln1_b, x1ln, x1lnbf, NN * TT);
  k_ffn<<<1500, 128, 0, stream>>>(x1lnbf, ff1p, ff1_b, ff2p, ff2_b, f);
  k_lnres<<<12000, 256, 0, stream>>>(x1ln, f, ln2_w, ln2_b, x2, nullptr, NN * TT);
  k_mean<<<(NN * DD + 255) / 256, 256, 0, stream>>>(x2, cbf);

  // ---- urban RGCN ----
  k_gemm<1, 2><<<dim3(125, 1), 128, 0, stream>>>(xubf, wsup, gbias_u, hu, DUU,
                                                 nullptr, 0, NN, 0, 0);
  k_gemm<1, 2><<<dim3(125, RR), 128, 0, stream>>>(xubf, wup, nullptr, proj, DUU,
                                                  nullptr, 0, NN, 1024, (long)NN * DUU);
  k_edge<<<8000, 256, 0, stream>>>(proj, src_urb, dst_urb, et_urb, hu, DUU, DUU);
  k_concat_u<<<(NN * DUU + 255) / 256, 256, 0, stream>>>(hu, cbf);

  // ---- global N x N attention head ----
  k_gemm<3, 2><<<dim3(125, 1), 128, 0, stream>>>(cbf, wqp, nullptr, q2f, 32,
                                                 q2bf, 32, NN, 0, 0);
  k_gemm<3, 2><<<dim3(125, 1), 128, 0, stream>>>(cbf, wkp, nullptr, k2, 32,
                                                 nullptr, 0, NN, 0, 0);
  k_gemm<3, 1><<<dim3(125, 1), 128, 0, stream>>>(cbf, wvp, nullptr, v2, 16,
                                                 nullptr, 0, NN, 0, 0);
  k_pack_b<<<(500 * 512 + 255) / 256, 256, 0, stream>>>(k2, k2p, 1, 500, 1);  // K^T
  k_pack_b<<<(250 * 512 + 255) / 256, 256, 0, stream>>>(v2, v2p, 250, 1, 0);
  k_gattn<<<500, 32, 0, stream>>>(q2bf, k2p, v2p, out_w, out_b, out);
}
